// GraphAttentionTemplate_78365973282924
// MI455X (gfx1250) — matrix operand order repaired and checked by reading, not compiled, not run on hardware
//
#include <hip/hip_runtime.h>
#include <hip/hip_bf16.h>
#include <math.h>

// -------- problem constants (match reference) --------
#define BATCHES 64
#define NDIM    512              // N == D == 512
#define TILE    128
#define KS      64               // K-step per staged tile (2 WMMA sub-steps of 32)
#define KPAD2   72               // LDS row pitch in halfs (144B: 16B-aligned, bank-spread)
#define NDND    ((long long)NDIM * NDIM)

typedef __attribute__((ext_vector_type(16))) _Float16 v16h;
typedef __attribute__((ext_vector_type(8)))  float    v8f;
typedef __attribute__((__vector_size__(4 * sizeof(int)))) int v4i;

union FragU  { v16h v; uint4 q[2]; };
union Pack8h { uint4 q; _Float16 h[8]; };
union Pack4f { uint4 q; float f[4]; };

// -------- CDNA5 async global->LDS staging (signature confirmed by probe) --------
#if defined(__has_builtin)
#if __has_builtin(__builtin_amdgcn_global_load_async_to_lds_b128) && \
    __has_builtin(__builtin_amdgcn_s_wait_asynccnt)
#define HAVE_ASYNC_LDS 1
#endif
#endif
#ifndef HAVE_ASYNC_LDS
#define HAVE_ASYNC_LDS 0
#endif

#if HAVE_ASYNC_LDS
typedef __attribute__((address_space(1))) v4i gas_v4i;   // global (AS1)
typedef __attribute__((address_space(3))) v4i lds_v4i;   // LDS    (AS3)
#endif

__device__ __forceinline__ void stage16B(const _Float16* __restrict__ g, _Float16* l) {
#if HAVE_ASYNC_LDS
    __builtin_amdgcn_global_load_async_to_lds_b128(
        (gas_v4i*)(void*)g, (lds_v4i*)l, 0, 0);
#else
    *(uint4*)l = *(const uint4*)g;
#endif
}

__device__ __forceinline__ void stage_fence() {
#if HAVE_ASYNC_LDS
    __builtin_amdgcn_s_wait_asynccnt(0);
#endif
}

// -------- f32 -> f16 (weights) --------
__global__ void f32_to_f16_kernel(const float* __restrict__ in,
                                  _Float16* __restrict__ out, int n) {
    int i = blockIdx.x * blockDim.x + threadIdx.x;
    if (i < n) out[i] = (_Float16)in[i];
}

// -------- batchinput [b][n][d] f32 -> Xt [b][d][n] f16 (tiled transpose) --------
__global__ __launch_bounds__(256)
void transpose_f32_to_f16t(const float* __restrict__ in, _Float16* __restrict__ out) {
    __shared__ float tile[32][33];
    const int b = blockIdx.z;
    const float* src = in  + (long long)b * NDND;
    _Float16*    dst = out + (long long)b * NDND;
    const int n0 = blockIdx.y * 32, d0 = blockIdx.x * 32;
    #pragma unroll
    for (int i = threadIdx.y; i < 32; i += 8)
        tile[i][threadIdx.x] = src[(long long)(n0 + i) * NDIM + d0 + threadIdx.x];
    __syncthreads();
    #pragma unroll
    for (int i = threadIdx.y; i < 32; i += 8)
        dst[(long long)(d0 + i) * NDIM + n0 + threadIdx.x] = (_Float16)tile[threadIdx.x][i];
}

// -------- generic 512^3 batched WMMA GEMM, "B-transposed" convention --------
// C[row][col] = sum_k A[row][k] * Bt[col][k]   (both operands K-contiguous f16)
// MODE 0: ChT[col][row]  = (f16)c                            (H -> stores Ht)
// MODE 1: Cf [col][row]  = c + biasCol[col]                  (T1 row-major via C=Ht@Wb^T)
// MODE 2: Cf [col][row]  = c + biasRow[row]                  (T2^T buffer)
// MODE 3: x = elu(c + biasCol[col]); OutF[row][col]=x; ChT[col][row]=(f16)x
template <int MODE>
__global__ __launch_bounds__(256)
void gemm512_wmma(const _Float16* __restrict__ A,  long long sA,
                  const _Float16* __restrict__ Bt, long long sB,
                  const float* __restrict__ biasRow,
                  const float* __restrict__ biasCol,
                  float* __restrict__ Cf,
                  _Float16* __restrict__ ChT,
                  float* __restrict__ OutF)
{
    __shared__ alignas(16) _Float16 As[2][TILE * KPAD2];
    __shared__ alignas(16) _Float16 Bs[2][TILE * KPAD2];

    const int b = blockIdx.z;
    A  += (long long)b * sA;
    Bt += (long long)b * sB;

    const int tid   = threadIdx.x;
    const int lane  = tid & 31;
    const int wave  = tid >> 5;         // 8 waves
    const int waveM = wave & 3;         // 4 along M -> 32 rows
    const int waveN = wave >> 2;        // 2 along N -> 64 cols
    const int blockRow = blockIdx.y * TILE;
    const int blockCol = blockIdx.x * TILE;

    const v8f zero = {0.f, 0.f, 0.f, 0.f, 0.f, 0.f, 0.f, 0.f};
    v8f acc[2][4];
    #pragma unroll
    for (int mi = 0; mi < 2; ++mi)
        #pragma unroll
        for (int ni = 0; ni < 4; ++ni)
            acc[mi][ni] = zero;

    // Stage a 128x64 A-tile and 128x64 Bt-tile into LDS buffer `bufi`.
    auto stage_tiles = [&](int bufi, int k0) {
        #pragma unroll
        for (int it = 0; it < 4; ++it) {
            int j  = tid + it * 256;             // 0..1023
            int r  = j >> 3;                     // 0..127
            int kb = (j & 7) << 3;               // 0..56
            stage16B(&A [(long long)(blockRow + r) * NDIM + k0 + kb],
                     &As[bufi][r * KPAD2 + kb]);
            stage16B(&Bt[(long long)(blockCol + r) * NDIM + k0 + kb],
                     &Bs[bufi][r * KPAD2 + kb]);
        }
    };

    stage_tiles(0, 0);                           // prefetch first tile

    int buf = 0;
    for (int k0 = 0; k0 < NDIM; k0 += KS, buf ^= 1) {
        stage_fence();                           // own asyncs done ...
        __syncthreads();                         // ... barrier makes it collective
        if (k0 + KS < NDIM)
            stage_tiles(buf ^ 1, k0 + KS);       // async copy overlaps compute below

        #pragma unroll
        for (int s = 0; s < KS; s += 32) {
            FragU afr[2], bfr[4];
            #pragma unroll
            for (int mi = 0; mi < 2; ++mi) {
                int row = waveM * 32 + mi * 16 + (lane & 15);
                int kb  = s + ((lane < 16) ? 0 : 8);
                afr[mi].q[0] = *(const uint4*)&As[buf][row * KPAD2 + kb];
                afr[mi].q[1] = *(const uint4*)&As[buf][row * KPAD2 + kb + 16];
            }
            #pragma unroll
            for (int ni = 0; ni < 4; ++ni) {
                int col = waveN * 64 + ni * 16 + (lane & 15);
                int kb  = s + ((lane < 16) ? 0 : 8);
                bfr[ni].q[0] = *(const uint4*)&Bs[buf][col * KPAD2 + kb];
                bfr[ni].q[1] = *(const uint4*)&Bs[buf][col * KPAD2 + kb + 16];
            }
            #pragma unroll
            for (int mi = 0; mi < 2; ++mi)
                #pragma unroll
                for (int ni = 0; ni < 4; ++ni)
                    acc[mi][ni] = __builtin_amdgcn_wmma_f32_16x16x32_f16(
                        false, afr[mi].v, false, bfr[ni].v,
                        (short)0, acc[mi][ni], false, false);
        }
        __syncthreads();                         // readers done before buf reuse
    }

    // ---- epilogue: per lane, rows v=0..7 are contiguous -> vector stores ----
    const int rbase = blockRow + waveM * 32;
    const int cbase = blockCol + waveN * 64;
    const int rOff  = (lane < 16) ? 0 : 8;
    const int cOff  = lane & 15;
    #pragma unroll
    for (int mi = 0; mi < 2; ++mi) {
        #pragma unroll
        for (int ni = 0; ni < 4; ++ni) {
            const int col  = cbase + ni * 16 + cOff;
            const int row0 = rbase + mi * 16 + rOff;
            if constexpr (MODE == 0) {
                Pack8h p;
                #pragma unroll
                for (int v = 0; v < 8; ++v) p.h[v] = (_Float16)acc[mi][ni][v];
                *(uint4*)&ChT[(long long)b * NDND + (long long)col * NDIM + row0] = p.q;
            } else if constexpr (MODE == 1) {
                const float bias = biasCol[col];
                Pack4f lo, hi;
                #pragma unroll
                for (int v = 0; v < 4; ++v) { lo.f[v] = acc[mi][ni][v] + bias;
                                              hi.f[v] = acc[mi][ni][v + 4] + bias; }
                float* dst = &Cf[(long long)b * NDND + (long long)col * NDIM + row0];
                *(uint4*)dst = lo.q; *(uint4*)(dst + 4) = hi.q;
            } else if constexpr (MODE == 2) {
                Pack4f lo, hi;
                #pragma unroll
                for (int v = 0; v < 4; ++v) { lo.f[v] = acc[mi][ni][v] + biasRow[row0 + v];
                                              hi.f[v] = acc[mi][ni][v + 4] + biasRow[row0 + 4 + v]; }
                float* dst = &Cf[(long long)b * NDND + (long long)col * NDIM + row0];
                *(uint4*)dst = lo.q; *(uint4*)(dst + 4) = hi.q;
            } else {
                const float bias = biasCol[col];
                Pack8h p;
                #pragma unroll
                for (int v = 0; v < 8; ++v) {
                    float x = acc[mi][ni][v] + bias;
                    x = x > 0.0f ? x : (expf(x) - 1.0f);      // ELU(alpha=1)
                    OutF[(long long)b * NDND + (long long)(row0 + v) * NDIM + col] = x;
                    p.h[v] = (_Float16)x;
                }
                *(uint4*)&ChT[(long long)b * NDND + (long long)col * NDIM + row0] = p.q;
            }
        }
    }
}

// -------- attention scores + masked row softmax --------
__global__ __launch_bounds__(256)
void attn_softmax_kernel(const float* __restrict__ T1,
                         const float* __restrict__ T2t,
                         const int* __restrict__ blen,
                         _Float16* __restrict__ Ph)
{
    const int x = blockIdx.x;
    const int b = blockIdx.y;
    const long long base = ((long long)b * NDIM + x) * NDIM;
    const int len = blen[b];
    const float scale = 0.04419417382415922f;   // 512^-0.5

    const int j0 = threadIdx.x, j1 = threadIdx.x + 256;
    float v0 = (T1[base + j0] + T2t[base + j0]) * scale;
    v0 = v0 >= 0.f ? v0 : 0.2f * v0;
    float s0 = (j0 < len) ? v0 : -INFINITY;
    float v1 = (T1[base + j1] + T2t[base + j1]) * scale;
    v1 = v1 >= 0.f ? v1 : 0.2f * v1;
    float s1 = (j1 < len) ? v1 : -INFINITY;

    __shared__ float red[256];
    red[threadIdx.x] = fmaxf(s0, s1);
    __syncthreads();
    for (int off = 128; off > 0; off >>= 1) {
        if (threadIdx.x < off)
            red[threadIdx.x] = fmaxf(red[threadIdx.x], red[threadIdx.x + off]);
        __syncthreads();
    }
    const float m = red[0];
    __syncthreads();

    const float e0 = expf(s0 - m);
    const float e1 = expf(s1 - m);
    red[threadIdx.x] = e0 + e1;
    __syncthreads();
    for (int off = 128; off > 0; off >>= 1) {
        if (threadIdx.x < off)
            red[threadIdx.x] += red[threadIdx.x + off];
        __syncthreads();
    }
    const float inv = 1.0f / red[0];
    Ph[base + j0] = (_Float16)(e0 * inv);
    Ph[base + j1] = (_Float16)(e1 * inv);
}

// -------- host orchestration --------
extern "C" void kernel_launch(void* const* d_in, const int* in_sizes, int n_in,
                              void* d_out, int out_size, void* d_ws, size_t ws_size,
                              hipStream_t stream)
{
    (void)in_sizes; (void)n_in; (void)out_size; (void)ws_size;
    const float* batchinput = (const float*)d_in[0];
    const int*   blen       = (const int*)d_in[1];
    const float* Wa         = (const float*)d_in[2];
    const float* Wb         = (const float*)d_in[3];
    const float* bb         = (const float*)d_in[4];
    const float* Wc         = (const float*)d_in[5];
    const float* bc         = (const float*)d_in[6];
    const float* bias       = (const float*)d_in[7];
    float* out = (float*)d_out;

    char* ws = (char*)d_ws;
    size_t off = 0;
    auto carve = [&](size_t bytes) -> char* {
        char* p = ws + off;
        off = (off + bytes + 255) & ~(size_t)255;
        return p;
    };
    _Float16* Wa16 = (_Float16*)carve((size_t)NDND * 2);
    _Float16* Wb16 = (_Float16*)carve((size_t)NDND * 2);
    _Float16* Wc16 = (_Float16*)carve((size_t)NDND * 2);
    _Float16* Xt   = (_Float16*)carve((size_t)BATCHES * NDND * 2);  // [b][d][n]
    _Float16* Ht   = (_Float16*)carve((size_t)BATCHES * NDND * 2);  // [b][d][m]
    _Float16* Ph   = (_Float16*)carve((size_t)BATCHES * NDND * 2);  // [b][i][j] row-major
    float*    T1   = (float*)   carve((size_t)BATCHES * NDND * 4);  // [b][x][y] row-major
    float*    T2t  = (float*)   carve((size_t)BATCHES * NDND * 4);  // [b][x][y] = T2[y][x]

    const int nW = NDIM * NDIM;
    f32_to_f16_kernel<<<(nW + 255) / 256, 256, 0, stream>>>(Wa, Wa16, nW);
    f32_to_f16_kernel<<<(nW + 255) / 256, 256, 0, stream>>>(Wb, Wb16, nW);
    f32_to_f16_kernel<<<(nW + 255) / 256, 256, 0, stream>>>(Wc, Wc16, nW);
    transpose_f32_to_f16t<<<dim3(16, 16, BATCHES), dim3(32, 8), 0, stream>>>(batchinput, Xt);

    const long long ND = NDND;
    const dim3 g(NDIM / TILE, NDIM / TILE, BATCHES);   // (4,4,64)
    const dim3 blk(256);
    for (int step = 0; step < 3; ++step) {
        // H = Wa @ X  (A=Wa row-major, Bt=Xt) -> Ht (f16, transposed store)
        gemm512_wmma<0><<<g, blk, 0, stream>>>(Wa16, 0, Xt, ND,
                                               nullptr, nullptr, nullptr, Ht, nullptr);
        // T1 row-major: C = Ht @ Wb^T (A=Ht, Bt=Wb16), +bb[col=o], transposed store
        gemm512_wmma<1><<<g, blk, 0, stream>>>(Ht, ND, Wb16, 0,
                                               nullptr, bb, T1, nullptr, nullptr);
        // T2^T buffer: C = Wc @ H (A=Wc16, Bt=Ht), +bc[row=o], transposed store
        gemm512_wmma<2><<<g, blk, 0, stream>>>(Wc16, 0, Ht, ND,
                                               bc, nullptr, T2t, nullptr, nullptr);
        // P = softmax(leaky_relu((T1 + T2^T) * scale) + mask)
        attn_softmax_kernel<<<dim3(NDIM, BATCHES), blk, 0, stream>>>(T1, T2t, blen, Ph);
        // out = elu(P @ H + bias[col=d]) ; Xt (f16, transposed) for next step
        gemm512_wmma<3><<<g, blk, 0, stream>>>(Ph, ND, Ht, ND,
                                               nullptr, bias, nullptr, Xt, out);
    }
}
